// LSGA_32590211842139
// MI455X (gfx1250) — compile-verified, matrix operand order checked by reading, not run on hardware
//
#include <hip/hip_runtime.h>
#include <math.h>

typedef __attribute__((ext_vector_type(16))) _Float16 v16h;
typedef __attribute__((ext_vector_type(8)))  _Float16 v8h;
typedef __attribute__((ext_vector_type(8)))  float    v8f;

#define CC   128   // channels
#define NN   8192  // points
#define KKN  16    // neighbors
#define H1D  32    // mlp hidden
#define BB   2

// ---- WMMA helpers (CDNA5 f16 16x16x32, f32 accumulate) --------------------
// 16-bit A-matrix 16x32 layout: lane L holds row M = L%16; with h = L/16,
// K elements { K0+8h .. K0+8h+7 } and { K0+16+8h .. K0+16+8h+7 }  (two b128
// LDS loads). B (32x16) mirrors this with column N = L%16 when the tile is
// stored row-per-column ([n][k]) — i.e. W row-major for   row @ W^T.
__device__ __forceinline__ v16h lds_frag(const _Float16* base, int stride,
                                         int K0, int lane) {
  const int m = lane & 15;
  const int h = lane >> 4;
  const _Float16* p = base + m * stride + K0 + 8 * h;
  v8h lo = *(const v8h*)(p);
  v8h hi = *(const v8h*)(p + 16);
  return __builtin_shufflevector(lo, hi, 0,1,2,3,4,5,6,7,8,9,10,11,12,13,14,15);
}

__device__ __forceinline__ v8f wmma32(v16h a, v16h b, v8f c) {
  return __builtin_amdgcn_wmma_f32_16x16x32_f16(false, a, false, b,
                                                (short)0, c, false, false);
}

// ---- Kernel Q: Q[(b*N+n)*C + o] = sum_c Wq[o][c]*xs[b][c][n] + bq[o] ------
__global__ __launch_bounds__(128) void lsga_q(const float* __restrict__ xs,
                                              const float* __restrict__ Wq,
                                              const float* __restrict__ bq,
                                              float* __restrict__ Qws) {
  __shared__ __align__(16) _Float16 lW[CC * CC];     // [o][c] == B tiles
  __shared__ __align__(16) _Float16 lA[4][16 * CC];  // per-wave A tile
  const int tid = threadIdx.x, lane = tid & 31, w = tid >> 5;
  for (int e = tid; e < CC * CC; e += 128) lW[e] = (_Float16)Wq[e];
  __syncthreads();

  const int row0 = (blockIdx.x * 4 + w) * 16;  // row = b*N + n
  const int b = row0 >> 13;
  const int n0 = row0 & (NN - 1);
  _Float16* A = &lA[w][0];
  for (int e = lane; e < 16 * CC; e += 32) {   // c-major: coalesced over n
    int m = e & 15, c = e >> 4;
    A[m * CC + c] = (_Float16)xs[((size_t)b * CC + c) * NN + n0 + m];
  }
  __syncthreads();

  // cache the 4 A-fragments: A-stationary across all 8 N-tiles
  v16h afr[4];
  for (int kk = 0; kk < 4; ++kk) afr[kk] = lds_frag(A, CC, kk * 32, lane);

  const int mcol = lane & 15, hh = lane >> 4;
  for (int t = 0; t < 8; ++t) {
    v8f acc = {};
    for (int kk = 0; kk < 4; ++kk)
      acc = wmma32(afr[kk], lds_frag(&lW[(t * 16) * CC], CC, kk * 32, lane),
                   acc);
    const int col = t * 16 + mcol;
    const float bias = bq[col];
    for (int r = 0; r < 8; ++r) {
      int m = r + 8 * hh;
      Qws[((size_t)(row0 + m)) * CC + col] = acc[r] + bias;
    }
  }
}

// ---- Main fused kernel: geo-MLP + gather + K/V proj + attention -----------
#define WAVES 4
#define PTS   2
__global__ __launch_bounds__(128) void lsga_main(
    const float* __restrict__ xs, const float* __restrict__ coords,
    const int* __restrict__ idx, const float* __restrict__ Bg,
    const float* __restrict__ W1, const float* __restrict__ b1,
    const float* __restrict__ W2, const float* __restrict__ b2,
    const float* __restrict__ Wk, const float* __restrict__ bk,
    const float* __restrict__ Wv, const float* __restrict__ bv,
    const float* __restrict__ Qws, float* __restrict__ outp) {
  __shared__ __align__(16) _Float16 lW1T[H1D * CC];  // W1^T : [32][128]
  __shared__ __align__(16) _Float16 lW2T[CC * H1D];  // W2^T : [128][32]
  __shared__ __align__(16) _Float16 lWk[CC * CC];    // Wk row-major
  __shared__ __align__(16) _Float16 lWv[CC * CC];
  __shared__ __align__(16) _Float16 lEf[WAVES][16 * CC];  // Efreq, reused as KVin
  __shared__ __align__(16) _Float16 lH1[WAVES][16 * H1D];
  __shared__ __align__(16) _Float16 lKt[WAVES][16 * CC];
  __shared__ __align__(16) _Float16 lVt[WAVES][16 * CC];
  __shared__ int lIdx[WAVES][16];

  const int tid = threadIdx.x, lane = tid & 31, w = tid >> 5;
  // Stage weights (f32 -> f16), transposing W1/W2 so B tiles are [n][k].
  for (int e = tid; e < H1D * CC; e += 128) {
    int j = e / CC, i = e % CC;                  // lW1T[j][i] = W1[i][j]
    lW1T[e] = (_Float16)W1[i * H1D + j];
  }
  for (int e = tid; e < CC * H1D; e += 128) {
    int c = e / H1D, j = e % H1D;                // lW2T[c][j] = W2[j][c]
    lW2T[e] = (_Float16)W2[j * CC + c];
  }
  for (int e = tid; e < CC * CC; e += 128) {
    lWk[e] = (_Float16)Wk[e];
    lWv[e] = (_Float16)Wv[e];
  }
  __syncthreads();

  const int blocksPerB = NN / (WAVES * PTS);
  const int b = blockIdx.x / blocksPerB;
  const int nbase = (blockIdx.x % blocksPerB) * (WAVES * PTS);
  const int mcol = lane & 15, hh = lane >> 4;
  const float TWO_PI = 6.28318530717958647f;

  for (int p = 0; p < PTS; ++p) {
    const int n = nbase + w * PTS + p;
    if (lane < 16) {
      int ii = idx[((size_t)b * NN + n) * KKN + lane];
      lIdx[w][lane] = ii;
      // global_prefetch_b8: warm neighbor coord lines for the Fourier stage
      __builtin_prefetch(&coords[((size_t)b * NN + ii) * 4], 0, 0);
    }
    // warm the Q row this wave will need after the GEMMs
    __builtin_prefetch(&Qws[((size_t)b * NN + n) * CC + lane * 4], 0, 0);
    __syncthreads();

    // ---- random-Fourier embedding: E_freq = [sin(2pi proj), cos(2pi proj)]
    const size_t cbase = ((size_t)b * NN + n) * 4;
    const float c0 = coords[cbase + 0], c1 = coords[cbase + 1];
    const float c2 = coords[cbase + 2], c3 = coords[cbase + 3];
    for (int k = 0; k < 16; ++k) {
      const size_t nb = ((size_t)b * NN + lIdx[w][k]) * 4;
      const float d0 = coords[nb + 0] - c0, d1 = coords[nb + 1] - c1;
      const float d2 = coords[nb + 2] - c2, d3 = coords[nb + 3] - c3;
      for (int jj = 0; jj < 2; ++jj) {
        const int j = lane + 32 * jj;  // j in 0..63
        float pr = d0 * Bg[j] + d1 * Bg[64 + j] + d2 * Bg[128 + j] +
                   d3 * Bg[192 + j];
        float ang = TWO_PI * pr;
        lEf[w][k * CC + j]      = (_Float16)__sinf(ang);
        lEf[w][k * CC + 64 + j] = (_Float16)__cosf(ang);
      }
    }
    __syncthreads();

    // ---- G1: h1 = relu(Efreq @ W1 + b1)   (16x128 @ 128x32)
    {
      v16h aEf[4];
      for (int kk = 0; kk < 4; ++kk)
        aEf[kk] = lds_frag(&lEf[w][0], CC, kk * 32, lane);
      for (int t = 0; t < 2; ++t) {
        v8f acc = {};
        for (int kk = 0; kk < 4; ++kk)
          acc = wmma32(aEf[kk],
                       lds_frag(&lW1T[(t * 16) * CC], CC, kk * 32, lane), acc);
        const int col = t * 16 + mcol;
        const float bias = b1[col];
        for (int r = 0; r < 8; ++r) {
          float v = acc[r] + bias;
          lH1[w][(r + 8 * hh) * H1D + col] = (_Float16)(v > 0.f ? v : 0.f);
        }
      }
    }
    __syncthreads();

    // ---- G2: kvin = h1 @ W2 + b2 + gather(xs)   (16x32 @ 32x128)
    {
      v16h aH = lds_frag(&lH1[w][0], H1D, 0, lane);  // single K-step, cached
      for (int t = 0; t < 8; ++t) {
        v16h bf = lds_frag(&lW2T[(t * 16) * H1D], H1D, 0, lane);
        v8f z = {};
        v8f acc = wmma32(aH, bf, z);
        const int col = t * 16 + mcol;
        const float bias = b2[col];
        for (int r = 0; r < 8; ++r) {
          int k = r + 8 * hh;
          float g = xs[((size_t)b * CC + col) * NN + lIdx[w][k]];
          lEf[w][k * CC + col] = (_Float16)(acc[r] + bias + g);  // reuse lEf
        }
      }
    }
    __syncthreads();

    // ---- G3/G4: Kt = kvin @ Wk^T + bk, Vt = kvin @ Wv^T + bv
    // A-stationary: one cached A-fragment set feeds 2 GEMMs x 8 N-tiles.
    {
      v16h aKV[4];
      for (int kk = 0; kk < 4; ++kk)
        aKV[kk] = lds_frag(&lEf[w][0], CC, kk * 32, lane);
      for (int t = 0; t < 8; ++t) {
        v8f aK = {}, aV = {};
        for (int kk = 0; kk < 4; ++kk) {
          v16h bK = lds_frag(&lWk[(t * 16) * CC], CC, kk * 32, lane);
          v16h bV = lds_frag(&lWv[(t * 16) * CC], CC, kk * 32, lane);
          aK = wmma32(aKV[kk], bK, aK);
          aV = wmma32(aKV[kk], bV, aV);
        }
        const int col = t * 16 + mcol;
        const float bkc = bk[col], bvc = bv[col];
        for (int r = 0; r < 8; ++r) {
          int k = r + 8 * hh;
          lKt[w][k * CC + col] = (_Float16)(aK[r] + bkc);
          lVt[w][k * CC + col] = (_Float16)(aV[r] + bvc);
        }
      }
    }
    __syncthreads();

    // ---- attention over 16 neighbors (per-wave, shfl reductions)
    float q[4];
    for (int i = 0; i < 4; ++i)
      q[i] = Qws[((size_t)b * NN + n) * CC + lane + 32 * i];
    float sc[16];
    for (int k = 0; k < 16; ++k) {
      float s = 0.f;
      for (int i = 0; i < 4; ++i)
        s += q[i] * (float)lKt[w][k * CC + lane + 32 * i];
      for (int off = 16; off; off >>= 1) s += __shfl_xor(s, off, 32);
      sc[k] = s * 0.08838834764831845f;  // 1/sqrt(128)
    }
    float mx = sc[0];
    for (int k = 1; k < 16; ++k) mx = fmaxf(mx, sc[k]);
    float sum = 0.f;
    for (int k = 0; k < 16; ++k) { sc[k] = __expf(sc[k] - mx); sum += sc[k]; }
    const float inv = 1.f / sum;
    for (int i = 0; i < 4; ++i) {
      const int c = lane + 32 * i;
      float o = 0.f;
      for (int k = 0; k < 16; ++k) o += sc[k] * (float)lVt[w][k * CC + c];
      outp[((size_t)b * NN + n) * CC + c] = o * inv;
    }
    __syncthreads();  // protect buffers before next point iteration
  }
}

// ---- Kernel O: out[b][o][n] = sum_c Wo[o][c]*outp[(b,n)][c] + bo[o] -------
__global__ __launch_bounds__(128) void lsga_o(const float* __restrict__ outp,
                                              const float* __restrict__ Wo,
                                              const float* __restrict__ bo,
                                              float* __restrict__ out) {
  __shared__ __align__(16) _Float16 lW[CC * CC];
  __shared__ __align__(16) _Float16 lA[4][16 * CC];
  const int tid = threadIdx.x, lane = tid & 31, w = tid >> 5;
  for (int e = tid; e < CC * CC; e += 128) lW[e] = (_Float16)Wo[e];
  __syncthreads();

  const int row0 = (blockIdx.x * 4 + w) * 16;
  const int b = row0 >> 13;
  const int n0 = row0 & (NN - 1);
  _Float16* A = &lA[w][0];
  for (int e = lane; e < 16 * CC; e += 32) {   // outp rows contiguous: coalesced
    int m = e >> 7, c = e & 127;
    A[m * CC + c] = (_Float16)outp[((size_t)row0 + m) * CC + c];
  }
  __syncthreads();

  v16h afr[4];
  for (int kk = 0; kk < 4; ++kk) afr[kk] = lds_frag(A, CC, kk * 32, lane);

  const int mcol = lane & 15, hh = lane >> 4;
  for (int t = 0; t < 8; ++t) {
    v8f acc = {};
    for (int kk = 0; kk < 4; ++kk)
      acc = wmma32(afr[kk], lds_frag(&lW[(t * 16) * CC], CC, kk * 32, lane),
                   acc);
    const int col = t * 16 + mcol;
    const float bias = bo[col];
    for (int r = 0; r < 8; ++r) {
      int m = r + 8 * hh;
      out[((size_t)b * CC + col) * NN + n0 + m] = acc[r] + bias;
    }
  }
}

extern "C" void kernel_launch(void* const* d_in, const int* in_sizes, int n_in,
                              void* d_out, int out_size, void* d_ws,
                              size_t ws_size, hipStream_t stream) {
  (void)in_sizes; (void)n_in; (void)out_size; (void)ws_size;
  const float* x      = (const float*)d_in[0];
  const float* coords = (const float*)d_in[1];
  const int*   idx    = (const int*)d_in[2];
  const float* Bg     = (const float*)d_in[3];
  const float* W1 = (const float*)d_in[4];
  const float* b1 = (const float*)d_in[5];
  const float* W2 = (const float*)d_in[6];
  const float* b2 = (const float*)d_in[7];
  const float* Wq = (const float*)d_in[8];
  const float* bq = (const float*)d_in[9];
  const float* Wk = (const float*)d_in[10];
  const float* bk = (const float*)d_in[11];
  const float* Wv = (const float*)d_in[12];
  const float* bv = (const float*)d_in[13];
  const float* Wo = (const float*)d_in[14];
  const float* bo = (const float*)d_in[15];

  float* Qws  = (float*)d_ws;                        // [B*N, C] f32 (8 MB)
  float* outp = Qws + (size_t)BB * NN * CC;          // [B*N, C] f32 (8 MB)
  float* out  = (float*)d_out;                       // [B, C, N] f32

  // 1) Q projection (1024 row-tiles of 16, 4 waves/WG)
  lsga_q<<<256, 128, 0, stream>>>(x, Wq, bq, Qws);
  // 2) fused Fourier-MLP + gather + K/V projection + attention
  lsga_main<<<BB * NN / (WAVES * PTS), 128, 0, stream>>>(
      x, coords, idx, Bg, W1, b1, W2, b2, Wk, bk, Wv, bv, Qws, outp);
  // 3) output projection, stored transposed to [B, C, N]
  lsga_o<<<256, 128, 0, stream>>>(outp, Wo, bo, out);
}
